// WeatherAwareGNN_944892805487
// MI455X (gfx1250) — matrix operand order repaired and checked by reading, not compiled, not run on hardware
//
#include <hip/hip_runtime.h>
#include <hip/hip_bf16.h>
#include <math.h>

typedef __attribute__((ext_vector_type(16))) __bf16 v16bf;
typedef __attribute__((ext_vector_type(8)))  float  v8f;

#define HIDN 256
#define HDN  64

// ============================================================================
// WMMA GEMM: C[M,N] = epi(A[M,K] @ B[K,N] + bias)
// A,B,bias,C are f32 in memory; tiles converted to bf16 in LDS; f32 accumulate.
// Block tile 128x64, 8 waves of 32 lanes, each wave a 32x32 macro-tile
// (2x2 v_wmma_f32_16x16x32_bf16 accumulators), K staged 64 per LDS fill.
// K must be a multiple of 32 (tail 32 zero-padded in LDS).
// EPI: 0 = none, 1 = relu, 2 = skip-mix: sk*(x)+(1-sk)*skipbuf, sk=sigmoid(*skp)
// ============================================================================
template<int EPI>
__global__ __launch_bounds__(256) void gemm_bf16_wmma(
    const float* __restrict__ A, const float* __restrict__ B,
    const float* __restrict__ bias, float* __restrict__ C,
    int M, int N, int K,
    const float* __restrict__ skipbuf, const float* __restrict__ skp)
{
  __shared__ __bf16 sA[128][72];   // [m][k], stride 144B keeps 16B alignment
  __shared__ __bf16 sBt[64][72];   // [n][k] (transposed for contiguous K reads)
  const int tid  = threadIdx.x;
  const int lane = tid & 31;
  const int wave = tid >> 5;
  const int wm   = wave & 3;       // 4 row-groups of 32
  const int wn   = wave >> 2;      // 2 col-groups of 32
  const int bRow = blockIdx.x * 128;
  const int bCol = blockIdx.y * 64;
  const bool fullM = (bRow + 128 <= M);

  v8f acc[2][2];
  #pragma unroll
  for (int ti = 0; ti < 2; ++ti)
    #pragma unroll
    for (int tj = 0; tj < 2; ++tj)
      #pragma unroll
      for (int v = 0; v < 8; ++v) acc[ti][tj][v] = 0.f;

  for (int k0 = 0; k0 < K; k0 += 64) {
    // ---- A tile: 128x64 f32 -> bf16, b128 global loads (8 float4/thread) ----
    #pragma unroll
    for (int j = 0; j < 8; ++j) {
      int idx = tid + j * 256;
      int r = idx >> 4, cc = (idx & 15) << 2;
      int gr = bRow + r;
      float4 v = make_float4(0.f, 0.f, 0.f, 0.f);
      if ((fullM || gr < M) && (k0 + cc < K))
        v = *(const float4*)(A + (size_t)gr * K + k0 + cc);
      sA[r][cc + 0] = (__bf16)v.x;
      sA[r][cc + 1] = (__bf16)v.y;
      sA[r][cc + 2] = (__bf16)v.z;
      sA[r][cc + 3] = (__bf16)v.w;
    }
    // ---- B tile: 64x64, b128 global loads (4 float4/thread), transposed -----
    #pragma unroll
    for (int j = 0; j < 4; ++j) {
      int idx = tid + j * 256;
      int kk = idx >> 4, n4 = (idx & 15) << 2;
      float4 v = make_float4(0.f, 0.f, 0.f, 0.f);
      if (k0 + kk < K)
        v = *(const float4*)(B + (size_t)(k0 + kk) * N + bCol + n4);
      sBt[n4 + 0][kk] = (__bf16)v.x;
      sBt[n4 + 1][kk] = (__bf16)v.y;
      sBt[n4 + 2][kk] = (__bf16)v.z;
      sBt[n4 + 3][kk] = (__bf16)v.w;
    }
    __syncthreads();

    const int khalf = lane >> 4;     // lane half selects K sub-ranges
    const int mrow  = lane & 15;
    #pragma unroll
    for (int ks = 0; ks < 64; ks += 32) {
      // A frag: lanes 0-15 hold K 0..7 & 16..23; lanes 16-31: 8..15 & 24..31
      v16bf afr[2];
      #pragma unroll
      for (int ti = 0; ti < 2; ++ti) {
        const __bf16* ap = &sA[wm * 32 + ti * 16 + mrow][ks + khalf * 8];
        #pragma unroll
        for (int q = 0; q < 8; ++q) afr[ti][q] = ap[q];
        #pragma unroll
        for (int q = 0; q < 8; ++q) afr[ti][8 + q] = ap[16 + q];
      }
      // B frag: lanes 0-15 hold K 0..7 & 16..23; lanes 16-31: 8..15 & 24..31
      v16bf bfr[2];
      #pragma unroll
      for (int tj = 0; tj < 2; ++tj) {
        const __bf16* bp = &sBt[wn * 32 + tj * 16 + mrow][ks + khalf * 8];
        #pragma unroll
        for (int q = 0; q < 8; ++q) bfr[tj][q] = bp[q];
        #pragma unroll
        for (int q = 0; q < 8; ++q) bfr[tj][8 + q] = bp[16 + q];
      }
      #pragma unroll
      for (int ti = 0; ti < 2; ++ti)
        #pragma unroll
        for (int tj = 0; tj < 2; ++tj)
          acc[ti][tj] = __builtin_amdgcn_wmma_f32_16x16x32_bf16(
              false, afr[ti], false, bfr[tj], (short)0, acc[ti][tj], false, false);
    }
    __syncthreads();
  }

  float sk = 1.f, osk = 0.f;
  if (EPI == 2) { sk = 1.f / (1.f + __expf(-skp[0])); osk = 1.f - sk; }
  const int nq = lane & 15;
  const int mh = (lane >> 4) << 3;   // C/D: M = vgpr + 8*(lane>=16)
  #pragma unroll
  for (int ti = 0; ti < 2; ++ti)
    #pragma unroll
    for (int tj = 0; tj < 2; ++tj) {
      int col  = bCol + wn * 32 + tj * 16 + nq;
      int row0 = bRow + wm * 32 + ti * 16 + mh;
      float bb = bias[col];
      size_t ofs = (size_t)row0 * N + col;     // strength-reduced addressing
      #pragma unroll
      for (int v = 0; v < 8; ++v) {
        if (fullM || (row0 + v < M)) {
          float x = acc[ti][tj][v] + bb;
          if (EPI == 1) x = fmaxf(x, 0.f);
          if (EPI == 2) x = sk * x + osk * skipbuf[ofs];
          C[ofs] = x;
        }
        ofs += (size_t)N;
      }
    }
}

// ============================================================================
// Fuse per-head rel transform into projection weights:
// out[c][h*64+e] = sum_d W[c][h*64+d] * rel[h][d][e]   (row 256 = fused bias)
// ============================================================================
__global__ void fuse_rel(const float* __restrict__ W, const float* __restrict__ bvec,
                         const float* __restrict__ rel, float* __restrict__ out)
{
  int c = blockIdx.x;            // 0..256 (256 = bias row)
  int oe = threadIdx.x;          // 0..255
  int h = oe >> 6, e = oe & 63;
  const float* wrow = (c < 256) ? (W + (size_t)c * 256) : bvec;
  const float* relh = rel + h * 4096;
  float acc = 0.f;
  #pragma unroll 8
  for (int d = 0; d < 64; ++d) acc += wrow[h * 64 + d] * relh[d * 64 + e];
  out[(size_t)c * 256 + oe] = acc;
}

// ---------------- ordered-uint float max encoding for atomicMax -------------
__device__ __forceinline__ unsigned fenc(float f) {
  unsigned u = __float_as_uint(f);
  return (u & 0x80000000u) ? ~u : (u | 0x80000000u);
}
__device__ __forceinline__ float fdec(unsigned e) {
  return (e & 0x80000000u) ? __uint_as_float(e & 0x7fffffffu) : __uint_as_float(~e);
}

// pass 1: alpha + atomic max per (dst, head); q/krel rows gathered as float4
__global__ void edge_alpha(const int* __restrict__ src, const int* __restrict__ dst,
                           const float* __restrict__ q, const float* __restrict__ krel,
                           const float* __restrict__ prel, float* __restrict__ exbuf,
                           unsigned* __restrict__ amax, int E)
{
  int idx = blockIdx.x * blockDim.x + threadIdx.x;
  if (idx >= E * 4) return;
  int e = idx >> 2, h = idx & 3;
  const float4* qp = (const float4*)(q    + (size_t)dst[e] * HIDN + h * HDN);
  const float4* kp = (const float4*)(krel + (size_t)src[e] * HIDN + h * HDN);
  float acc = 0.f;
  #pragma unroll 4
  for (int i = 0; i < HDN / 4; ++i) {
    float4 a = qp[i], b = kp[i];
    acc += a.x * b.x + a.y * b.y + a.z * b.z + a.w * b.w;
  }
  float a = acc * prel[h] * 0.125f;   // * p_rel * 1/sqrt(64)
  exbuf[idx] = a;
  atomicMax(&amax[(size_t)dst[e] * 4 + h], fenc(a));
}

// pass 2: exp(alpha - max), atomic denominator
__global__ void edge_expsum(const int* __restrict__ dst, float* __restrict__ exbuf,
                            const unsigned* __restrict__ amax, float* __restrict__ den, int E)
{
  int idx = blockIdx.x * blockDim.x + threadIdx.x;
  if (idx >= E * 4) return;
  int e = idx >> 2, h = idx & 3;
  float m = fdec(amax[(size_t)dst[e] * 4 + h]);
  float x = __expf(exbuf[idx] - m);
  exbuf[idx] = x;
  atomicAdd(&den[(size_t)dst[e] * 4 + h], x);
}

// pass 3: weighted scatter-aggregate (one block per edge, 256 channels)
__global__ void edge_agg(const int* __restrict__ src, const int* __restrict__ dst,
                         const float* __restrict__ exbuf, const float* __restrict__ den,
                         const float* __restrict__ vrel, float* __restrict__ agg, int E)
{
  long long idx = (long long)blockIdx.x * 256 + threadIdx.x;
  if (idx >= (long long)E * HIDN) return;
  int e = (int)(idx >> 8), c = (int)(idx & 255), h = c >> 6;
  float w = exbuf[(size_t)e * 4 + h] / (den[(size_t)dst[e] * 4 + h] + 1e-16f);
  atomicAdd(&agg[(size_t)dst[e] * HIDN + c], w * vrel[(size_t)src[e] * HIDN + c]);
}

__global__ void fill_u32(unsigned* __restrict__ p, unsigned v, long long n)
{
  long long i = (long long)blockIdx.x * 256 + threadIdx.x;
  if (i < n) p[i] = v;
}

__global__ void gelu_ip(float* __restrict__ x, long long n)
{
  long long i = (long long)blockIdx.x * 256 + threadIdx.x;
  if (i < n) { float v = x[i]; x[i] = 0.5f * v * (1.f + erff(v * 0.70710678118654752f)); }
}

// player nodes receive no edges: h = sk*a_b + (1-sk)*h
__global__ void player_skip(float* __restrict__ hp, const float* __restrict__ ab,
                            const float* __restrict__ skp, long long n)
{
  long long i = (long long)blockIdx.x * 256 + threadIdx.x;
  if (i < n) {
    float sk = 1.f / (1.f + __expf(-skp[0]));
    hp[i] = sk * ab[i & 255] + (1.f - sk) * hp[i];
  }
}

__global__ void concat2(const float* __restrict__ a, int ca, const float* __restrict__ b,
                        int cb, float* __restrict__ o, long long n)
{
  long long i = (long long)blockIdx.x * 256 + threadIdx.x;
  if (i >= n) return;
  int w = ca + cb; long long r = i / w; int c = (int)(i % w);
  o[i] = (c < ca) ? a[r * ca + c] : b[r * cb + (c - ca)];
}

// ------------------------------- encoders -----------------------------------
__global__ void weather_enc(const float* __restrict__ wf,
    const float* tw, const float* tb, const float* hw, const float* hb,
    const float* ww, const float* wb, const float* pw, const float* pb,
    const float* f1w, const float* f1b, const float* f2w, const float* f2b,
    float* __restrict__ we)
{
  int m = blockIdx.x, t = threadIdx.x;          // 64 threads
  __shared__ float cat[64], hid[64];
  const float* w = wf + (size_t)m * 6;
  float v;
  if (t < 16)       v = w[0] * tw[t] + tb[t];
  else if (t < 32)  { int j = t - 16; v = w[1] * hw[j] + hb[j]; }
  else if (t < 48)  { int j = t - 32; v = w[2] * ww[j] + w[3] * ww[16 + j] + wb[j]; }
  else              { int j = t - 48; v = w[4] * pw[j] + w[5] * pw[16 + j] + pb[j]; }
  cat[t] = v;
  __syncthreads();
  float a = 0.f;
  for (int i = 0; i < 64; ++i) a += cat[i] * f1w[i * 64 + t];
  hid[t] = fmaxf(a + f1b[t], 0.f);
  __syncthreads();
  float b = 0.f;
  for (int i = 0; i < 64; ++i) b += hid[i] * f2w[i * 64 + t];
  we[(size_t)m * 64 + t] = b + f2b[t];
}

__global__ void coord_enc(const float* __restrict__ coords,
    const float* la1w, const float* la1b, const float* la2w, const float* la2b,
    const float* lo1w, const float* lo1b, const float* lo2w, const float* lo2b,
    const float* f1w, const float* f1b, const float* f2w, const float* f2b,
    float* __restrict__ ce)
{
  int v = blockIdx.x, t = threadIdx.x;          // 32 threads
  __shared__ float tA[16], tB[16], cat[32], hid[32];
  float lat = coords[(size_t)v * 2 + 0] / 90.f;
  float lon = coords[(size_t)v * 2 + 1] / 180.f;
  if (t < 16) tA[t] = fmaxf(lat * la1w[t] + la1b[t], 0.f);
  else        { int j = t - 16; tB[j] = fmaxf(lon * lo1w[j] + lo1b[j], 0.f); }
  __syncthreads();
  float a = 0.f;
  if (t < 16) { for (int i = 0; i < 16; ++i) a += tA[i] * la2w[i * 16 + t]; a += la2b[t]; }
  else        { int j = t - 16; for (int i = 0; i < 16; ++i) a += tB[i] * lo2w[i * 16 + j]; a += lo2b[j]; }
  cat[t] = a;
  __syncthreads();
  float h = 0.f;
  for (int i = 0; i < 32; ++i) h += cat[i] * f1w[i * 32 + t];
  hid[t] = fmaxf(h + f1b[t], 0.f);
  __syncthreads();
  float o = 0.f;
  for (int i = 0; i < 32; ++i) o += hid[i] * f2w[i * 32 + t];
  ce[(size_t)v * 32 + t] = o + f2b[t];
}

__global__ void squad_enc(const int* __restrict__ squad,
    const float* role, const float* bat, const float* bowl,
    const float* f1w, const float* f1b, const float* f2w, const float* f2b,
    float* __restrict__ se)
{
  int p = blockIdx.x, t = threadIdx.x;          // 64 threads (48 active)
  __shared__ float cat[28], hid[48];
  const int* sq = squad + (size_t)p * 3;
  int s0 = min(max(sq[0], 0), 4), s1 = min(max(sq[1], 0), 2), s2 = min(max(sq[2], 0), 4);
  if (t < 12)      cat[t] = role[s0 * 12 + t];
  else if (t < 20) cat[t] = bat[s1 * 8 + (t - 12)];
  else if (t < 28) cat[t] = bowl[s2 * 8 + (t - 20)];
  __syncthreads();
  if (t < 48) {
    float a = 0.f;
    for (int i = 0; i < 28; ++i) a += cat[i] * f1w[i * 48 + t];
    hid[t] = fmaxf(a + f1b[t], 0.f);
  }
  __syncthreads();
  if (t < 48) {
    float b = 0.f;
    for (int i = 0; i < 48; ++i) b += hid[i] * f2w[i * 48 + t];
    se[(size_t)p * 48 + t] = b + f2b[t];
  }
}

__global__ void impact_head(const float* __restrict__ we, const float* __restrict__ ce,
    const float* w1, const float* b1, const float* w2, const float* b2,
    const float* w3, const float* b3, float* __restrict__ imp)
{
  int m = blockIdx.x, t = threadIdx.x;          // 64 threads
  __shared__ float wc[96], h1[64], h2[32];
  wc[t] = we[(size_t)m * 64 + t];
  if (t < 32) wc[64 + t] = ce[(size_t)m * 32 + t];
  __syncthreads();
  float a = 0.f;
  for (int i = 0; i < 96; ++i) a += wc[i] * w1[i * 64 + t];
  h1[t] = fmaxf(a + b1[t], 0.f);
  __syncthreads();
  if (t < 32) {
    float b = 0.f;
    for (int i = 0; i < 64; ++i) b += h1[i] * w2[i * 32 + t];
    h2[t] = fmaxf(b + b2[t], 0.f);
  }
  __syncthreads();
  if (t == 0) {
    float c = 0.f;
    for (int i = 0; i < 32; ++i) c += h2[i] * w3[i];
    imp[m] = 1.f / (1.f + __expf(-(c + b3[0])));
  }
}

// ============================================================================
// Host orchestration
// ============================================================================
static void run_gemm(int epi, const float* A, const float* B, const float* bias,
                     float* C, int M, int N, int K,
                     const float* skipbuf, const float* skp, hipStream_t s)
{
  dim3 g((unsigned)((M + 127) / 128), (unsigned)(N / 64)), b(256);
  if (epi == 0)      gemm_bf16_wmma<0><<<g, b, 0, s>>>(A, B, bias, C, M, N, K, nullptr, nullptr);
  else if (epi == 1) gemm_bf16_wmma<1><<<g, b, 0, s>>>(A, B, bias, C, M, N, K, nullptr, nullptr);
  else               gemm_bf16_wmma<2><<<g, b, 0, s>>>(A, B, bias, C, M, N, K, skipbuf, skp);
}

static void zfill(void* p, long long n_u32, hipStream_t s)
{
  fill_u32<<<(unsigned)((n_u32 + 255) / 256), 256, 0, s>>>((unsigned*)p, 0u, n_u32);
}

extern "C" void kernel_launch(void* const* d_in, const int* in_sizes, int n_in,
                              void* d_out, int out_size, void* d_ws, size_t ws_size,
                              hipStream_t stream)
{
  (void)out_size;
  const float* player_x = (const float*)d_in[0];
  const float* venue_x  = (const float*)d_in[1];
  const float* match_x  = (const float*)d_in[2];
  const float* weather  = (const float*)d_in[3];
  const float* coords   = (const float*)d_in[4];
  const int*   squad    = (const int*)d_in[5];
  const int*   eipv     = (const int*)d_in[6];
  const int*   eipm     = (const int*)d_in[7];
  const int*   eivm     = (const int*)d_in[8];

  const int NP  = in_sizes[0] / 128;
  const int NV  = in_sizes[1] / 64;
  const int NM  = in_sizes[2] / 96;
  const int EPV = in_sizes[6] / 2;
  const int EPM = in_sizes[7] / 2;
  const int EVM = in_sizes[8] / 2;
  const int NX  = (NV > NM) ? NV : NM;

  // ---- param leaf table in setup_inputs() insertion order ----
  int ps[160]; int npar = 0;
  {
    auto add = [&](int s) { ps[npar++] = s; };
    // weather: tw tb hw hb ww wb pw pb f1w f1b f2w f2b
    add(16); add(16); add(16); add(16); add(32); add(16); add(32); add(16);
    add(4096); add(64); add(4096); add(64);
    // coord: la1w la1b la2w la2b lo1w lo1b lo2w lo2b f1w f1b f2w f2b
    add(16); add(16); add(256); add(16); add(16); add(16); add(256); add(16);
    add(1024); add(32); add(1024); add(32);
    // squad: role bat bowl f1w f1b f2w f2b
    add(60); add(24); add(72); add(1344); add(48); add(2304); add(48);
    // proj: pw pb vw vb mw mb
    add(32768); add(256); add(24576); add(256); add(40960); add(256);
    // hgt: 3 layers x [ (k,q,v,a)x(w,b) + skip per type; (a_rel,m_rel,p_rel) per et ]
    for (int l = 0; l < 3; ++l) {
      for (int t = 0; t < 3; ++t) { for (int k = 0; k < 4; ++k) { add(65536); add(256); } add(1); }
      for (int e = 0; e < 3; ++e) { add(16384); add(16384); add(4); }
    }
    // out: w1 b1 w2 b2
    add(32768); add(128); add(16384); add(128);
    // impact: w1 b1 w2 b2 w3 b3
    add(6144); add(64); add(2048); add(32); add(32); add(1);
  }
  const float* P[160];
  if (n_in >= 9 + npar) {
    for (int i = 0; i < npar; ++i) P[i] = (const float*)d_in[9 + i];
  } else {  // params passed as one concatenated buffer
    const float* base = (const float*)d_in[9];
    long long off = 0;
    for (int i = 0; i < npar; ++i) { P[i] = base + off; off += ps[i]; }
  }
#define LP(l, o) P[37 + (l) * 36 + (o)]

  // ---- workspace carve-out ----
  float* Wb = (float*)d_ws;
  size_t woff = 0;
  auto alloc = [&](size_t n) -> float* {
    float* p = Wb + woff; woff += (n + 255) & ~(size_t)255; return p;
  };
  float* we    = alloc((size_t)NM * 64);
  float* ce    = alloc((size_t)NV * 32);
  float* vx    = alloc((size_t)NV * 96);
  float* mx    = alloc((size_t)NM * 160);
  float* hp    = alloc((size_t)NP * 256);
  float* hv    = alloc((size_t)NV * 256);
  float* hm    = alloc((size_t)NM * 256);
  float* krelA = alloc((size_t)NP * 256);
  float* vrelA = alloc((size_t)NP * 256);
  float* krelB = alloc((size_t)NV * 256);
  float* vrelB = alloc((size_t)NV * 256);
  float* qbuf  = alloc((size_t)NX * 256);
  float* aggV  = alloc((size_t)NV * 256);
  float* aggM  = alloc((size_t)NM * 256);
  unsigned* amax = (unsigned*)alloc((size_t)NX * 4);
  float* den   = alloc((size_t)NX * 4);
  float* ex    = alloc((size_t)(EPM + EVM) * 4);
  float* fw    = alloc((size_t)6 * 257 * 256);
  float* tmp   = krelA;  // reused for output-MLP hidden
  if (ws_size < woff * sizeof(float)) return;  // insufficient scratch

  float* out   = (float*)d_out;
  float* emb_p = out;
  float* emb_v = emb_p + (size_t)NP * 128;
  float* emb_m = emb_v + (size_t)NV * 128;
  float* imp   = emb_m + (size_t)NM * 128;
  float* se    = imp + NM;

  // ---- encoders ----
  weather_enc<<<NM, 64, 0, stream>>>(weather, P[0], P[1], P[2], P[3], P[4], P[5],
                                     P[6], P[7], P[8], P[9], P[10], P[11], we);
  coord_enc<<<NV, 32, 0, stream>>>(coords, P[12], P[13], P[14], P[15], P[16], P[17],
                                   P[18], P[19], P[20], P[21], P[22], P[23], ce);
  squad_enc<<<NP, 64, 0, stream>>>(squad, P[24], P[25], P[26], P[27], P[28], P[29], P[30], se);

  // ---- input projections ----
  {
    long long nvx = (long long)NV * 96, nmx = (long long)NM * 160;
    concat2<<<(unsigned)((nvx + 255) / 256), 256, 0, stream>>>(venue_x, 64, ce, 32, vx, nvx);
    concat2<<<(unsigned)((nmx + 255) / 256), 256, 0, stream>>>(match_x, 96, we, 64, mx, nmx);
  }
  run_gemm(0, player_x, P[31], P[32], hp, NP, 256, 128, 0, 0, stream);
  run_gemm(0, vx,       P[33], P[34], hv, NV, 256,  96, 0, 0, stream);
  run_gemm(0, mx,       P[35], P[36], hm, NM, 256, 160, 0, 0, stream);

  // ---- 3 HGT layers ----
  const size_t FS = 257 * 256;
  float *fKpv = fw, *fVpv = fw + FS, *fKpm = fw + 2 * FS,
        *fVpm = fw + 3 * FS, *fKvm = fw + 4 * FS, *fVvm = fw + 5 * FS;
  for (int l = 0; l < 3; ++l) {
    // fold a_rel/m_rel into K/V projection weights (+ fused bias row)
    fuse_rel<<<257, 256, 0, stream>>>(LP(l, 0), LP(l, 1), LP(l, 27), fKpv);  // k player, a_rel_pv
    fuse_rel<<<257, 256, 0, stream>>>(LP(l, 4), LP(l, 5), LP(l, 28), fVpv);  // v player, m_rel_pv
    fuse_rel<<<257, 256, 0, stream>>>(LP(l, 0), LP(l, 1), LP(l, 30), fKpm);
    fuse_rel<<<257, 256, 0, stream>>>(LP(l, 4), LP(l, 5), LP(l, 31), fVpm);
    fuse_rel<<<257, 256, 0, stream>>>(LP(l, 9), LP(l, 10), LP(l, 33), fKvm); // k venue, a_rel_vm
    fuse_rel<<<257, 256, 0, stream>>>(LP(l, 13), LP(l, 14), LP(l, 34), fVvm);

    // ---- venue destination (pv edges) ----
    run_gemm(0, hp, fKpv, fKpv + 65536, krelA, NP, 256, 256, 0, 0, stream);
    run_gemm(0, hp, fVpv, fVpv + 65536, vrelA, NP, 256, 256, 0, 0, stream);
    run_gemm(0, hv, LP(l, 11), LP(l, 12), qbuf, NV, 256, 256, 0, 0, stream); // q venue
    zfill(amax, (long long)NV * 4, stream);
    zfill(den,  (long long)NV * 4, stream);
    zfill(aggV, (long long)NV * 256, stream);
    {
      unsigned gb = (unsigned)(((long long)EPV * 4 + 255) / 256);
      edge_alpha<<<gb, 256, 0, stream>>>(eipv, eipv + EPV, qbuf, krelA, LP(l, 29), ex, amax, EPV);
      edge_expsum<<<gb, 256, 0, stream>>>(eipv + EPV, ex, amax, den, EPV);
      edge_agg<<<(unsigned)EPV, 256, 0, stream>>>(eipv, eipv + EPV, ex, den, vrelA, aggV, EPV);
    }
    gelu_ip<<<(unsigned)(((long long)NV * 256 + 255) / 256), 256, 0, stream>>>(aggV, (long long)NV * 256);

    // ---- match destination (pm + vm edges, joint softmax) ----
    run_gemm(0, hp, fKpm, fKpm + 65536, krelA, NP, 256, 256, 0, 0, stream);
    run_gemm(0, hp, fVpm, fVpm + 65536, vrelA, NP, 256, 256, 0, 0, stream);
    run_gemm(0, hv, fKvm, fKvm + 65536, krelB, NV, 256, 256, 0, 0, stream);
    run_gemm(0, hv, fVvm, fVvm + 65536, vrelB, NV, 256, 256, 0, 0, stream);
    run_gemm(0, hm, LP(l, 20), LP(l, 21), qbuf, NM, 256, 256, 0, 0, stream); // q match
    zfill(amax, (long long)NM * 4, stream);
    zfill(den,  (long long)NM * 4, stream);
    zfill(aggM, (long long)NM * 256, stream);
    {
      float* exvm = ex + (size_t)EPM * 4;
      unsigned gp = (unsigned)(((long long)EPM * 4 + 255) / 256);
      unsigned gv = (unsigned)(((long long)EVM * 4 + 255) / 256);
      edge_alpha<<<gp, 256, 0, stream>>>(eipm, eipm + EPM, qbuf, krelA, LP(l, 32), ex, amax, EPM);
      edge_alpha<<<gv, 256, 0, stream>>>(eivm, eivm + EVM, qbuf, krelB, LP(l, 35), exvm, amax, EVM);
      edge_expsum<<<gp, 256, 0, stream>>>(eipm + EPM, ex, amax, den, EPM);
      edge_expsum<<<gv, 256, 0, stream>>>(eivm + EVM, exvm, amax, den, EVM);
      edge_agg<<<(unsigned)EPM, 256, 0, stream>>>(eipm, eipm + EPM, ex, den, vrelA, aggM, EPM);
      edge_agg<<<(unsigned)EVM, 256, 0, stream>>>(eivm, eivm + EVM, exvm, den, vrelB, aggM, EVM);
    }
    gelu_ip<<<(unsigned)(((long long)NM * 256 + 255) / 256), 256, 0, stream>>>(aggM, (long long)NM * 256);

    // ---- gated updates ----
    run_gemm(2, aggV, LP(l, 15), LP(l, 16), hv, NV, 256, 256, hv, LP(l, 17), stream);
    run_gemm(2, aggM, LP(l, 24), LP(l, 25), hm, NM, 256, 256, hm, LP(l, 26), stream);
    player_skip<<<(unsigned)(((long long)NP * 256 + 255) / 256), 256, 0, stream>>>(
        hp, LP(l, 7), LP(l, 8), (long long)NP * 256);
  }

  // ---- output MLPs ----
  const float *ow1 = P[145], *ob1 = P[146], *ow2 = P[147], *ob2 = P[148];
  run_gemm(1, hp, ow1, ob1, tmp, NP, 128, 256, 0, 0, stream);
  run_gemm(0, tmp, ow2, ob2, emb_p, NP, 128, 128, 0, 0, stream);
  run_gemm(1, hv, ow1, ob1, tmp, NV, 128, 256, 0, 0, stream);
  run_gemm(0, tmp, ow2, ob2, emb_v, NV, 128, 128, 0, 0, stream);
  run_gemm(1, hm, ow1, ob1, tmp, NM, 128, 256, 0, 0, stream);
  run_gemm(0, tmp, ow2, ob2, emb_m, NM, 128, 128, 0, 0, stream);

  // ---- impact head ----
  impact_head<<<NM, 64, 0, stream>>>(we, ce, P[149], P[150], P[151], P[152], P[153], P[154], imp);
}
